// Model_65755949301917
// MI455X (gfx1250) — matrix operand order repaired and checked by reading, not compiled, not run on hardware
//
#include <hip/hip_runtime.h>
#include <hip/hip_bf16.h>
#include <math.h>

typedef __attribute__((ext_vector_type(16))) _Float16 v16h;
typedef __attribute__((ext_vector_type(8)))  float    v8f;

#define BSZ   4096
#define LX    1000
#define MM    16
#define KK    20
#define KJ    80      // 4*KK  (reduction length)
#define KJP   96      // padded to 3 WMMA K-tiles of 32
#define TOUT  981     // LX-KK+1
#define NTILE 62      // ceil(981/16)
#define PRIOR_LOC   (-0.01f)
#define PRIOR_SCALE (0.001f)
#define NEG_BIG     (-3.0e38f)

// ---------------------------------------------------------------------------
// Weight prep: softplus, two rsamples, f16 pack (K padded 80->96), KL sum, s.
// ---------------------------------------------------------------------------
__global__ __launch_bounds__(256) void prep_weights(
    const float* __restrict__ qw_loc, const float* __restrict__ qw_scale_raw,
    const float* __restrict__ gen_scale,
    const float* __restrict__ eps1, const float* __restrict__ eps2,
    _Float16* __restrict__ wh1, _Float16* __restrict__ wh2,
    _Float16* __restrict__ whloc, float* __restrict__ klsum,
    float* __restrict__ sval) {
  __shared__ float red[256];
  const int tid = threadIdx.x;
  float local = 0.0f;
  for (int i = tid; i < MM * KJ; i += 256) {
    const int m = i / KJ, j = i - m * KJ;
    const float loc = qw_loc[i];
    const float raw = qw_scale_raw[i];
    const float qs  = (raw > 20.0f) ? raw : log1pf(expf(raw));
    wh1[m * KJP + j]   = (_Float16)(loc + qs * eps1[i]);
    wh2[m * KJP + j]   = (_Float16)(loc + qs * eps2[i]);
    whloc[m * KJP + j] = (_Float16)loc;
    const float dl = loc - PRIOR_LOC;
    local += logf(PRIOR_SCALE / qs)
           + (qs * qs + dl * dl) / (2.0f * PRIOR_SCALE * PRIOR_SCALE) - 0.5f;
  }
  // zero the K-padding (j in [80,96))
  if (tid < MM * (KJP - KJ)) {
    const int m = tid / (KJP - KJ), j = KJ + (tid % (KJP - KJ));
    wh1[m * KJP + j] = (_Float16)0.0f;
    wh2[m * KJP + j] = (_Float16)0.0f;
    whloc[m * KJP + j] = (_Float16)0.0f;
  }
  red[tid] = local;
  __syncthreads();
  for (int s = 128; s > 0; s >>= 1) {
    if (tid < s) red[tid] += red[tid + s];
    __syncthreads();
  }
  if (tid == 0) {
    *klsum = red[0];
    const float g  = gen_scale[0];
    const float sp = (g > 20.0f) ? g : log1pf(expf(g));
    *sval = 1.0f / sp;
  }
}

// ---------------------------------------------------------------------------
// Motif conv as implicit GEMM on v_wmma_f32_16x16x32_f16, fused max+sigmoid.
// One workgroup (8 wave32) per batch sample.  p[b,m] = sigmoid(max_t score).
// ---------------------------------------------------------------------------
__global__ __launch_bounds__(256) void motif_conv_wmma(
    const float* __restrict__ x, const _Float16* __restrict__ wh,
    float* __restrict__ pout) {
  __shared__ _Float16 xs[4 * 1024];     // per-channel stride 1024, zero padded
  __shared__ float partial[8][MM];
  const int b   = blockIdx.x;
  const int tid = threadIdx.x;

  for (int i = tid; i < 4 * 1024; i += 256) xs[i] = (_Float16)0.0f;
  __syncthreads();
  const float* xb = x + (size_t)b * (4 * LX);
  for (int i = tid; i < 4 * LX; i += 256) {
    const int c = i / LX, pos = i - c * LX;
    xs[c * 1024 + pos] = (_Float16)xb[i];
  }
  __syncthreads();

  const int wave = tid >> 5;
  const int lane = tid & 31;
  const int col  = lane & 15;   // motif column (B/D), time row group (A)
  const int hi   = lane >> 4;   // half-wave selector

  // B fragments: 32x16 tiles, column = motif, K contiguous per half-wave
  v16h bfrag[3];
  #pragma unroll
  for (int jt = 0; jt < 3; ++jt) {
    const int j0 = jt * 32 + hi * 16;
    #pragma unroll
    for (int h = 0; h < 16; ++h) bfrag[jt][h] = wh[col * KJP + j0 + h];
  }

  float mloc = NEG_BIG;
  for (int tile = wave; tile < NTILE; tile += 8) {
    const int t0 = tile * 16;
    v8f acc = {};
    #pragma unroll
    for (int jt = 0; jt < 3; ++jt) {
      // A fragment: row = time position t0+col; K contiguous per half-wave:
      // lanes 0-15: K in {0..15}; lanes 16-31: K in {16..31}
      v16h afrag;
      const int trow = t0 + col;
      #pragma unroll
      for (int h = 0; h < 16; ++h) {
        const int j = jt * 32 + hi * 16 + h;
        _Float16 v = (_Float16)0.0f;
        if (j < KJ) {
          const int c = j / KK, k = j - c * KK;
          v = xs[c * 1024 + trow + k];
        }
        afrag[h] = v;
      }
      acc = __builtin_amdgcn_wmma_f32_16x16x32_f16(
          false, afrag, false, bfrag[jt], (short)0, acc, false, false);
    }
    // fused max_t with tail masking (rows t>=981 invalid)
    #pragma unroll
    for (int i = 0; i < 8; ++i) {
      const int t = t0 + i + hi * 8;
      mloc = fmaxf(mloc, (t < TOUT) ? acc[i] : NEG_BIG);
    }
  }
  // lane l and l^16 hold complementary D rows for the same motif column
  mloc = fmaxf(mloc, __shfl_xor(mloc, 16, 32));
  if (hi == 0) partial[wave][col] = mloc;
  __syncthreads();
  if (tid < MM) {
    float mx = NEG_BIG;
    #pragma unroll
    for (int w = 0; w < 8; ++w) mx = fmaxf(mx, partial[w][tid]);
    pout[b * MM + tid] = 1.0f / (1.0f + expf(-mx));
  }
}

// ---------------------------------------------------------------------------
// BCE over batch per motif: bce[m] = -mean_b(y*clip(log p) + (1-y)*clip(log(1-p)))
// ---------------------------------------------------------------------------
__global__ __launch_bounds__(256) void bce_kernel(
    const float* __restrict__ p2, const float* __restrict__ y,
    float* __restrict__ bce) {
  __shared__ float red[256];
  const int m = blockIdx.x, tid = threadIdx.x;
  float acc = 0.0f;
  for (int b = tid; b < BSZ; b += 256) {
    const float p = p2[b * MM + m];
    const float lp  = fmaxf(logf(p), -100.0f);
    const float l1p = fmaxf(logf(1.0f - p), -100.0f);
    acc += y[b] * lp + (1.0f - y[b]) * l1p;
  }
  red[tid] = acc;
  __syncthreads();
  for (int s = 128; s > 0; s >>= 1) {
    if (tid < s) red[tid] += red[tid + s];
    __syncthreads();
  }
  if (tid == 0) bce[m] = -red[0] / (float)BSZ;
}

// ---------------------------------------------------------------------------
// Per-sample ELBO assembly + mean into loss (d_out[0], pre-zeroed).
// ---------------------------------------------------------------------------
__global__ __launch_bounds__(256) void loss_kernel(
    const float* __restrict__ p1, const float* __restrict__ p2,
    const float* __restrict__ bce, const float* __restrict__ klsum,
    const float* __restrict__ sval, float* __restrict__ loss) {
  __shared__ float red[256];
  const int b = blockIdx.x * 256 + threadIdx.x;
  const float s  = *sval;
  const float kl = *klsum;
  float lp[MM], ez[MM];
  float mx = NEG_BIG;
  #pragma unroll
  for (int m = 0; m < MM; ++m) {
    const float p = p1[b * MM + m];
    const float v = logf(p) + log1pf(-p);
    lp[m] = v;
    mx = fmaxf(mx, v);
  }
  float zsum = 0.0f;
  #pragma unroll
  for (int m = 0; m < MM; ++m) { ez[m] = expf(lp[m] - mx); zsum += ez[m]; }
  float ent = 0.0f, term = 0.0f;
  #pragma unroll
  for (int m = 0; m < MM; ++m) {
    const float zm = ez[m] / zsum;
    ent += zm * logf(1e-20f + zm);
    const float p = p2[b * MM + m];
    const float gll = -logf(1e-16f + powf(1e-16f + p, s)
                                   + powf(1e-16f + 1.0f - p, s)) - s * bce[m];
    term += zm * gll;
  }
  red[threadIdx.x] = ent + 2.0f * kl - term;
  __syncthreads();
  for (int st = 128; st > 0; st >>= 1) {
    if (threadIdx.x < st) red[threadIdx.x] += red[threadIdx.x + st];
    __syncthreads();
  }
  if (threadIdx.x == 0) atomicAdd(loss, red[0] / (float)BSZ);
}

// ---------------------------------------------------------------------------
// Predict: assign = argmax_m [log p_loc + log1p(-p_loc)]; softmax is monotone
// so argmax(lp) == argmax(softmax(lp)).  The gathered per-sample conv in the
// reference equals p_loc[b, assign] exactly — no extra conv needed.
// ---------------------------------------------------------------------------
__global__ __launch_bounds__(256) void predict_kernel(
    const float* __restrict__ ploc, float* __restrict__ ypred) {
  const int b = blockIdx.x * 256 + threadIdx.x;
  if (b >= BSZ) return;
  float best = NEG_BIG;
  int bi = 0;
  #pragma unroll
  for (int m = 0; m < MM; ++m) {
    const float p = ploc[b * MM + m];
    const float v = logf(p) + log1pf(-p);
    if (v > best) { best = v; bi = m; }   // first-max tie break, like argmax
  }
  ypred[b] = ploc[b * MM + bi];
}

// ---------------------------------------------------------------------------
extern "C" void kernel_launch(void* const* d_in, const int* in_sizes, int n_in,
                              void* d_out, int out_size, void* d_ws,
                              size_t ws_size, hipStream_t stream) {
  const float* x            = (const float*)d_in[0];  // [B,4,L]
  const float* y            = (const float*)d_in[1];  // [B]
  const float* qw_loc       = (const float*)d_in[2];  // [M,80]
  const float* qw_scale_raw = (const float*)d_in[3];  // [M,80]
  const float* gen_scale    = (const float*)d_in[4];  // [1]
  const float* eps1         = (const float*)d_in[5];  // [M,80]
  const float* eps2         = (const float*)d_in[6];  // [M,80]
  float* out = (float*)d_out;                         // [0]=loss, [1..B]=y_pred

  char* ws = (char*)d_ws;
  size_t off = 0;
  auto carve = [&](size_t bytes) {
    void* p = ws + off;
    off = (off + bytes + 255) & ~(size_t)255;
    return p;
  };
  float*    p1    = (float*)carve(BSZ * MM * sizeof(float));
  float*    p2    = (float*)carve(BSZ * MM * sizeof(float));
  float*    ploc  = (float*)carve(BSZ * MM * sizeof(float));
  _Float16* wh1   = (_Float16*)carve(MM * KJP * sizeof(_Float16));
  _Float16* wh2   = (_Float16*)carve(MM * KJP * sizeof(_Float16));
  _Float16* whloc = (_Float16*)carve(MM * KJP * sizeof(_Float16));
  float*    bce   = (float*)carve(MM * sizeof(float));
  float*    klsum = (float*)carve(sizeof(float));
  float*    sval  = (float*)carve(sizeof(float));

  hipMemsetAsync(out, 0, sizeof(float), stream);  // zero loss accumulator

  prep_weights<<<1, 256, 0, stream>>>(qw_loc, qw_scale_raw, gen_scale,
                                      eps1, eps2, wh1, wh2, whloc, klsum, sval);

  motif_conv_wmma<<<BSZ, 256, 0, stream>>>(x, wh1, p1);    // z-path sample
  motif_conv_wmma<<<BSZ, 256, 0, stream>>>(x, wh2, p2);    // generative sample
  motif_conv_wmma<<<BSZ, 256, 0, stream>>>(x, whloc, ploc);// mean weights

  bce_kernel<<<MM, 256, 0, stream>>>(p2, y, bce);
  loss_kernel<<<BSZ / 256, 256, 0, stream>>>(p1, p2, bce, klsum, sval, out);
  predict_kernel<<<BSZ / 256, 256, 0, stream>>>(ploc, out + 1);
}
